// RelPositionAttention_59270548685650
// MI455X (gfx1250) — compile-verified, matrix operand order checked by reading, not run on hardware
//
#include <hip/hip_runtime.h>

// ---------------------------------------------------------------------------
// RelPositionAttention (Transformer-XL style) for gfx1250 / MI455X.
// B=4, T=1024, D=1024, H=16, DK=64.  All matmuls via v_wmma_f32_16x16x32_bf16.
// HBM->LDS staging via GLOBAL_LOAD_ASYNC_TO_LDS_B128 (ASYNCcnt), double
// buffered in the K=1024 projection GEMMs.
// d_out = [ out (B*T*D fp32) | attn (B*H*T*T fp32) ]
// Workspace use: ~318 MB.
// ---------------------------------------------------------------------------

#define USE_ASYNC 1

typedef __bf16 bf16_t;
typedef __attribute__((ext_vector_type(16))) __bf16 v16bf;
typedef __attribute__((ext_vector_type(8)))  __bf16 v8bf;
typedef __attribute__((ext_vector_type(2)))  __bf16 v2bf;
typedef __attribute__((ext_vector_type(8)))  float  v8f;

#define CB 4
#define CT 1024
#define CD 1024
#define CH 16
#define CDK 64
#define LDS_STRIDE 40   // 32 + 8 pad (80B rows, 16B aligned)
#define LDS_STRIDE64 72 // 64 + 8 pad (144B rows, 16B aligned)

// --- CDNA5 async HBM->LDS copy, 16B per lane; tracked by ASYNCcnt.
__device__ inline void async_ld_b128(void* lds_dst, const void* gsrc) {
  unsigned l = (unsigned)(size_t)lds_dst;   // low 32 bits of flat LDS addr = LDS byte offset
  asm volatile("global_load_async_to_lds_b128 %0, %1, off"
               :: "v"(l), "v"(gsrc) : "memory");
}
__device__ inline void wait_async_le(int n) {
  if (n == 0) asm volatile("s_wait_asynccnt 0x0" ::: "memory");
  else        asm volatile("s_wait_asynccnt 0x4" ::: "memory");
}

// --- load a 16x32 bf16 WMMA operand fragment from an LDS tile laid out
// row-major [row][k] with row stride `stride`.
// Layout per CDNA5 ISA 7.12.2 (16-bit A 16x32): lanes 0-15 hold K={0..7,16..23},
// lanes 16-31 hold K={8..15,24..31} for rows 0..15.
__device__ inline v16bf load_frag(const bf16_t* lds, int row0, int stride) {
  const int lane = threadIdx.x & 31;
  const int r  = row0 + (lane & 15);
  const int kb = (lane >> 4) ? 8 : 0;
  const bf16_t* p = lds + r * stride + kb;
  v8bf lo = *(const v8bf*)(p);        // K = kb .. kb+7
  v8bf hi = *(const v8bf*)(p + 16);   // K = kb+16 .. kb+23
  v16bf f;
#pragma unroll
  for (int i = 0; i < 8; ++i) { f[i] = lo[i]; f[i + 8] = hi[i]; }
  return f;
}

// ---------------------------------------------------------------------------
// fp32 -> bf16 flat convert (weights, pos_emb)
// ---------------------------------------------------------------------------
__global__ void cvt_bf16_kernel(const float* __restrict__ in,
                                bf16_t* __restrict__ out, int n) {
  int i = blockIdx.x * blockDim.x + threadIdx.x;
  if (i < n) out[i] = (bf16_t)in[i];
}

// ---------------------------------------------------------------------------
// LayerNorm row (D=1024) -> bf16
// ---------------------------------------------------------------------------
__global__ __launch_bounds__(256)
void ln_bf16_kernel(const float* __restrict__ x, const float* __restrict__ g,
                    const float* __restrict__ bb, bf16_t* __restrict__ xb) {
  __shared__ float red[256];
  const size_t row = blockIdx.x;
  const float* xr = x + row * CD;
  const int t = threadIdx.x;
  float v[4]; float s = 0.f;
#pragma unroll
  for (int k = 0; k < 4; ++k) { v[k] = xr[t + k * 256]; s += v[k]; }
  red[t] = s; __syncthreads();
  for (int st = 128; st > 0; st >>= 1) { if (t < st) red[t] += red[t + st]; __syncthreads(); }
  const float mu = red[0] * (1.0f / CD); __syncthreads();
  float s2 = 0.f;
#pragma unroll
  for (int k = 0; k < 4; ++k) { float d = v[k] - mu; s2 += d * d; }
  red[t] = s2; __syncthreads();
  for (int st = 128; st > 0; st >>= 1) { if (t < st) red[t] += red[t + st]; __syncthreads(); }
  const float inv = rsqrtf(red[0] * (1.0f / CD) + 1e-5f);
#pragma unroll
  for (int k = 0; k < 4; ++k) {
    int j = t + k * 256;
    xb[row * CD + j] = (bf16_t)((v[k] - mu) * inv * g[j] + bb[j]);
  }
}

// ---------------------------------------------------------------------------
// NT GEMM: C[M,N] = A[M,1024] @ W[N,1024]^T, 128x128 block tile, 8 waves.
// Double-buffered async HBM->LDS staging; epilogues fuse bias / pos-bias /
// head-split layout.
// ---------------------------------------------------------------------------
enum { EPI_QUQV = 0, EPI_HEADS = 1, EPI_PH = 2, EPI_OUTPROJ = 3 };

template <int EPI>
__global__ __launch_bounds__(256)
void gemm_xw_kernel(const bf16_t* __restrict__ A, const bf16_t* __restrict__ W,
                    const float* __restrict__ bias,
                    bf16_t* __restrict__ out0, bf16_t* __restrict__ out1,
                    float* __restrict__ outf,
                    const float* __restrict__ pu, const float* __restrict__ pv) {
  const int tid = threadIdx.x;
  const int n0 = blockIdx.x * 128, m0 = blockIdx.y * 128;
  const int wid = tid >> 5, lane = tid & 31;
  const int wm = wid >> 1, wn = wid & 1;            // 4x2 wave grid, 32x64 / wave
  const int lr = tid >> 1, lc = (tid & 1) * 16;     // staging: 16 elems / thread
  v8f acc[2][4] = {};

#if USE_ASYNC
  __shared__ __attribute__((aligned(16))) bf16_t at[2][128 * LDS_STRIDE];
  __shared__ __attribute__((aligned(16))) bf16_t bt[2][128 * LDS_STRIDE];
  auto issue_tile = [&](int k0, int buf) {
    const bf16_t* ga = A + (size_t)(m0 + lr) * CD + k0 + lc;
    const bf16_t* gw = W + (size_t)(n0 + lr) * CD + k0 + lc;
    async_ld_b128(&at[buf][lr * LDS_STRIDE + lc],     ga);
    async_ld_b128(&at[buf][lr * LDS_STRIDE + lc + 8], ga + 8);
    async_ld_b128(&bt[buf][lr * LDS_STRIDE + lc],     gw);
    async_ld_b128(&bt[buf][lr * LDS_STRIDE + lc + 8], gw + 8);
  };
  issue_tile(0, 0);
  int cur = 0;
  for (int k0 = 0; k0 < CD; k0 += 32, cur ^= 1) {
    if (k0 + 32 < CD) { issue_tile(k0 + 32, cur ^ 1); wait_async_le(4); }
    else              { wait_async_le(0); }
    __syncthreads();
    v16bf af0 = load_frag(at[cur], wm * 32, LDS_STRIDE);
    v16bf af1 = load_frag(at[cur], wm * 32 + 16, LDS_STRIDE);
#pragma unroll
    for (int j = 0; j < 4; ++j) {
      v16bf bfj = load_frag(bt[cur], wn * 64 + j * 16, LDS_STRIDE);
      acc[0][j] = __builtin_amdgcn_wmma_f32_16x16x32_bf16(
          false, af0, false, bfj, (short)0, acc[0][j], false, false);
      acc[1][j] = __builtin_amdgcn_wmma_f32_16x16x32_bf16(
          false, af1, false, bfj, (short)0, acc[1][j], false, false);
    }
    __syncthreads();
  }
#else
  __shared__ __attribute__((aligned(16))) bf16_t at[128 * LDS_STRIDE];
  __shared__ __attribute__((aligned(16))) bf16_t bt[128 * LDS_STRIDE];
  for (int k0 = 0; k0 < CD; k0 += 32) {
    const bf16_t* ga = A + (size_t)(m0 + lr) * CD + k0 + lc;
    const bf16_t* gw = W + (size_t)(n0 + lr) * CD + k0 + lc;
    *(v8bf*)(at + lr * LDS_STRIDE + lc)     = *(const v8bf*)(ga);
    *(v8bf*)(at + lr * LDS_STRIDE + lc + 8) = *(const v8bf*)(ga + 8);
    *(v8bf*)(bt + lr * LDS_STRIDE + lc)     = *(const v8bf*)(gw);
    *(v8bf*)(bt + lr * LDS_STRIDE + lc + 8) = *(const v8bf*)(gw + 8);
    __syncthreads();
    v16bf af0 = load_frag(at, wm * 32, LDS_STRIDE);
    v16bf af1 = load_frag(at, wm * 32 + 16, LDS_STRIDE);
#pragma unroll
    for (int j = 0; j < 4; ++j) {
      v16bf bfj = load_frag(bt, wn * 64 + j * 16, LDS_STRIDE);
      acc[0][j] = __builtin_amdgcn_wmma_f32_16x16x32_bf16(
          false, af0, false, bfj, (short)0, acc[0][j], false, false);
      acc[1][j] = __builtin_amdgcn_wmma_f32_16x16x32_bf16(
          false, af1, false, bfj, (short)0, acc[1][j], false, false);
    }
    __syncthreads();
  }
#endif

  const int lhi = (lane >> 4) * 8, lnn = lane & 15;
#pragma unroll
  for (int i = 0; i < 2; ++i)
#pragma unroll
    for (int j = 0; j < 4; ++j)
#pragma unroll
      for (int e = 0; e < 8; ++e) {
        const int m = m0 + wm * 32 + i * 16 + lhi + e;
        const int n = n0 + wn * 64 + j * 16 + lnn;
        const float c = acc[i][j][e];
        if (EPI == EPI_QUQV) {
          const int b = m >> 10, t = m & (CT - 1);
          const int h = n >> 6,  dk = n & (CDK - 1);
          const size_t o = (((size_t)(b * CH + h)) * CT + t) * CDK + dk;
          const float base = c + bias[n];
          out0[o] = (bf16_t)(base + pu[n]);   // p_u flat [H*DK] == [n]
          out1[o] = (bf16_t)(base + pv[n]);
        } else if (EPI == EPI_HEADS) {
          const int b = m >> 10, t = m & (CT - 1);
          const int h = n >> 6,  dk = n & (CDK - 1);
          const size_t o = (((size_t)(b * CH + h)) * CT + t) * CDK + dk;
          out0[o] = (bf16_t)(c + bias[n]);
        } else if (EPI == EPI_PH) {           // M == T, layout [H,T,DK]
          const int h = n >> 6, dk = n & (CDK - 1);
          const size_t o = ((size_t)h * CT + m) * CDK + dk;
          out0[o] = (bf16_t)(c + bias[n]);
        } else {                              // EPI_OUTPROJ: fp32 [B*T, D]
          outf[(size_t)m * CD + n] = c + bias[n];
        }
      }
}

// ---------------------------------------------------------------------------
// Per-(b,h) score GEMM: S[T,T] = Q[T,64] @ X[T,64]^T  (X = K or P[h]).
// Whole K=64 panel staged once (async), single barrier, 16 WMMAs / wave.
// ---------------------------------------------------------------------------
template <int IS_BD>
__global__ __launch_bounds__(256)
void gemm_scores_kernel(const bf16_t* __restrict__ qmat,
                        const bf16_t* __restrict__ kmat,   // kh (IS_BD=0) or ph (IS_BD=1)
                        float* __restrict__ sbuf) {
  __shared__ __attribute__((aligned(16))) bf16_t at[128 * LDS_STRIDE64];
  __shared__ __attribute__((aligned(16))) bf16_t bt[128 * LDS_STRIDE64];
  const int tid = threadIdx.x;
  const int bh = blockIdx.z;
  const int n0 = blockIdx.x * 128, m0 = blockIdx.y * 128;
  const bf16_t* Ab = qmat + (size_t)bh * CT * CDK;
  const bf16_t* Bb = IS_BD ? (kmat + (size_t)(bh & (CH - 1)) * CT * CDK)
                           : (kmat + (size_t)bh * CT * CDK);
  float* Sb = sbuf + (size_t)bh * CT * CT;
  const int wid = tid >> 5, lane = tid & 31;
  const int wm = wid >> 1, wn = wid & 1;
  const int lr = tid >> 1, lc = (tid & 1) * 32;     // 32 elems / thread / matrix
  v8f acc[2][4] = {};

  {
    const bf16_t* ga = Ab + (size_t)(m0 + lr) * CDK + lc;
    const bf16_t* gb = Bb + (size_t)(n0 + lr) * CDK + lc;
#if USE_ASYNC
#pragma unroll
    for (int o = 0; o < 32; o += 8) {
      async_ld_b128(&at[lr * LDS_STRIDE64 + lc + o], ga + o);
      async_ld_b128(&bt[lr * LDS_STRIDE64 + lc + o], gb + o);
    }
    wait_async_le(0);
#else
#pragma unroll
    for (int o = 0; o < 32; o += 8) {
      *(v8bf*)(at + lr * LDS_STRIDE64 + lc + o) = *(const v8bf*)(ga + o);
      *(v8bf*)(bt + lr * LDS_STRIDE64 + lc + o) = *(const v8bf*)(gb + o);
    }
#endif
  }
  __syncthreads();

#pragma unroll
  for (int k0 = 0; k0 < CDK; k0 += 32) {
    v16bf af0 = load_frag(at + k0, wm * 32, LDS_STRIDE64);
    v16bf af1 = load_frag(at + k0, wm * 32 + 16, LDS_STRIDE64);
#pragma unroll
    for (int j = 0; j < 4; ++j) {
      v16bf bfj = load_frag(bt + k0, wn * 64 + j * 16, LDS_STRIDE64);
      acc[0][j] = __builtin_amdgcn_wmma_f32_16x16x32_bf16(
          false, af0, false, bfj, (short)0, acc[0][j], false, false);
      acc[1][j] = __builtin_amdgcn_wmma_f32_16x16x32_bf16(
          false, af1, false, bfj, (short)0, acc[1][j], false, false);
    }
  }

  const int lhi = (lane >> 4) * 8, lnn = lane & 15;
#pragma unroll
  for (int i = 0; i < 2; ++i)
#pragma unroll
    for (int j = 0; j < 4; ++j)
#pragma unroll
      for (int e = 0; e < 8; ++e) {
        const int m = m0 + wm * 32 + i * 16 + lhi + e;
        const int n = n0 + wn * 64 + j * 16 + lnn;
        Sb[(size_t)m * CT + n] = acc[i][j][e];
      }
}

// ---------------------------------------------------------------------------
// rel-shift + scale + mask + softmax; in-place on attn row (ac already there).
// shifted_bd[q,j] = j<=q ? bd[q, T-1-q+j] : (j==q+1 ? 0 : bd[q+1, j-q-2])
// ---------------------------------------------------------------------------
__global__ __launch_bounds__(256)
void softmax_shift_kernel(float* __restrict__ attn, const float* __restrict__ bd,
                          const unsigned char* __restrict__ mask) {
  __shared__ float red[256];
  const int q = blockIdx.x, bh = blockIdx.y, b = bh >> 4;
  float* row = attn + ((size_t)bh * CT + q) * CT;
  const float* bdb = bd + (size_t)bh * CT * CT;
  const unsigned char* mrow = mask + ((size_t)b * CT + q) * CT;
  const int t = threadIdx.x;
  float s[4];
#pragma unroll
  for (int k = 0; k < 4; ++k) {
    const int j = t + k * 256;
    float bdv;
    if (j <= q)           bdv = bdb[(size_t)q * CT + (CT - 1 - q + j)];
    else if (j == q + 1)  bdv = 0.f;
    else                  bdv = bdb[(size_t)(q + 1) * CT + (j - q - 2)];
    float v = (row[j] + bdv) * 0.125f;          // 1/sqrt(DK)
    if (mrow[j]) v = -__builtin_inff();
    s[k] = v;
  }
  float mx = fmaxf(fmaxf(s[0], s[1]), fmaxf(s[2], s[3]));
  red[t] = mx; __syncthreads();
  for (int st = 128; st > 0; st >>= 1) { if (t < st) red[t] = fmaxf(red[t], red[t + st]); __syncthreads(); }
  mx = red[0]; __syncthreads();
  float e[4], ssum = 0.f;
#pragma unroll
  for (int k = 0; k < 4; ++k) { e[k] = __expf(s[k] - mx); ssum += e[k]; }
  red[t] = ssum; __syncthreads();
  for (int st = 128; st > 0; st >>= 1) { if (t < st) red[t] += red[t + st]; __syncthreads(); }
  const float inv = 1.0f / red[0];
#pragma unroll
  for (int k = 0; k < 4; ++k) row[t + k * 256] = e[k] * inv;
}

// ---------------------------------------------------------------------------
// Per-(b,h) AV GEMM (NN): O[T,64] = attn[T,T] @ V[T,64]
// A converted fp32->bf16 pairwise during LDS staging (v_cvt_pk_bf16_f32 +
// ds_store_b32); V tile read as b128 and transposed into LDS; next A panel
// prefetched (global_prefetch_b8). Output merged-head bf16 [B,T,D].
// ---------------------------------------------------------------------------
__global__ __launch_bounds__(256)
void gemm_av_kernel(const float* __restrict__ attn, const bf16_t* __restrict__ vh,
                    bf16_t* __restrict__ ao) {
  __shared__ __attribute__((aligned(16))) bf16_t at[128 * LDS_STRIDE];
  __shared__ __attribute__((aligned(16))) bf16_t bt[64 * LDS_STRIDE];
  const int tid = threadIdx.x;
  const int bh = blockIdx.z, m0 = blockIdx.y * 128;
  const int b = bh >> 4, h = bh & (CH - 1);
  const float*  Ab = attn + (size_t)bh * CT * CT;
  const bf16_t* Vb = vh + (size_t)bh * CT * CDK;
  const int wid = tid >> 5, lane = tid & 31;
  const int wm = wid >> 1, wn = wid & 1;              // 4x2 waves, 32x32 / wave
  const int lr = tid >> 1, lc = (tid & 1) * 16;
  const int vk = tid >> 5, vn = (tid & 31) * 2;       // unused helper slot
  (void)vk; (void)vn;
  v8f acc[2][2] = {};

  for (int k0 = 0; k0 < CT; k0 += 32) {
    const float* ga = Ab + (size_t)(m0 + lr) * CT + k0 + lc;
    if (k0 + 32 < CT) __builtin_prefetch(ga + 32, 0, 1);   // next A panel
#pragma unroll
    for (int e = 0; e < 16; e += 2) {                 // packed cvt + 4B LDS store
      v2bf p; p[0] = (bf16_t)ga[e]; p[1] = (bf16_t)ga[e + 1];
      *(v2bf*)(at + lr * LDS_STRIDE + lc + e) = p;
    }
    {                                                 // B tile 32(k) x 64(n), transpose
      const int idx = tid * 8;
      const int kk = idx >> 6, nn = idx & (CDK - 1);
      v8bf vv = *(const v8bf*)(Vb + (size_t)(k0 + kk) * CDK + nn);
#pragma unroll
      for (int e = 0; e < 8; ++e) bt[(nn + e) * LDS_STRIDE + kk] = vv[e];
    }
    __syncthreads();
    v16bf af0 = load_frag(at, wm * 32, LDS_STRIDE);
    v16bf af1 = load_frag(at, wm * 32 + 16, LDS_STRIDE);
#pragma unroll
    for (int j = 0; j < 2; ++j) {
      v16bf bfj = load_frag(bt, wn * 32 + j * 16, LDS_STRIDE);
      acc[0][j] = __builtin_amdgcn_wmma_f32_16x16x32_bf16(
          false, af0, false, bfj, (short)0, acc[0][j], false, false);
      acc[1][j] = __builtin_amdgcn_wmma_f32_16x16x32_bf16(
          false, af1, false, bfj, (short)0, acc[1][j], false, false);
    }
    __syncthreads();
  }

  const int lhi = (lane >> 4) * 8, lnn = lane & 15;
#pragma unroll
  for (int i = 0; i < 2; ++i)
#pragma unroll
    for (int j = 0; j < 2; ++j)
#pragma unroll
      for (int e = 0; e < 8; ++e) {
        const int m = m0 + wm * 32 + i * 16 + lhi + e;       // t
        const int n = wn * 32 + j * 16 + lnn;                // dk
        ao[((size_t)b * CT + m) * CD + h * CDK + n] = (bf16_t)acc[i][j][e];
      }
}

// ---------------------------------------------------------------------------
extern "C" void kernel_launch(void* const* d_in, const int* in_sizes, int n_in,
                              void* d_out, int out_size, void* d_ws, size_t ws_size,
                              hipStream_t stream) {
  const float* query  = (const float*)d_in[0];
  const float* posemb = (const float*)d_in[1];
  const unsigned char* mask = (const unsigned char*)d_in[2];
  const float* ln_g = (const float*)d_in[3];
  const float* ln_b = (const float*)d_in[4];
  const float* Wq = (const float*)d_in[5];  const float* bq = (const float*)d_in[6];
  const float* Wk = (const float*)d_in[7];  const float* bk = (const float*)d_in[8];
  const float* Wv = (const float*)d_in[9];  const float* bv = (const float*)d_in[10];
  const float* Wp = (const float*)d_in[11]; const float* bp = (const float*)d_in[12];
  const float* Wo = (const float*)d_in[13]; const float* bo = (const float*)d_in[14];
  const float* p_u = (const float*)d_in[15];
  const float* p_v = (const float*)d_in[16];

  float* out_f  = (float*)d_out;                         // [B,T,D]
  float* attn_f = out_f + (size_t)CB * CT * CD;          // [B,H,T,T]

  char* w = (char*)d_ws;
  size_t off = 0;
  auto take = [&](size_t bytes) -> void* {
    void* p = w + off; off = (off + bytes + 255) & ~(size_t)255; return p;
  };
  const size_t NE_BTD = (size_t)CB * CT * CD;            // 4M elems
  const size_t NE_DD  = (size_t)CD * CD;                 // 1M elems
  bf16_t* xb  = (bf16_t*)take(NE_BTD * 2);
  bf16_t* wqb = (bf16_t*)take(NE_DD * 2);
  bf16_t* wkb = (bf16_t*)take(NE_DD * 2);
  bf16_t* wvb = (bf16_t*)take(NE_DD * 2);
  bf16_t* wpb = (bf16_t*)take(NE_DD * 2);
  bf16_t* wob = (bf16_t*)take(NE_DD * 2);
  bf16_t* peb = (bf16_t*)take((size_t)CT * CD * 2);
  bf16_t* qu  = (bf16_t*)take(NE_BTD * 2);
  bf16_t* qv  = (bf16_t*)take(NE_BTD * 2);
  bf16_t* kh  = (bf16_t*)take(NE_BTD * 2);
  bf16_t* vh  = (bf16_t*)take(NE_BTD * 2);
  bf16_t* ph  = (bf16_t*)take((size_t)CH * CT * CDK * 2);
  bf16_t* ao  = (bf16_t*)take(NE_BTD * 2);
  float*  bdb = (float*)take((size_t)CB * CH * CT * CT * 4);   // 256 MB

  // 1) converts
  const int cvtBlocks = (int)((NE_DD + 255) / 256);
  cvt_bf16_kernel<<<cvtBlocks, 256, 0, stream>>>(Wq, wqb, (int)NE_DD);
  cvt_bf16_kernel<<<cvtBlocks, 256, 0, stream>>>(Wk, wkb, (int)NE_DD);
  cvt_bf16_kernel<<<cvtBlocks, 256, 0, stream>>>(Wv, wvb, (int)NE_DD);
  cvt_bf16_kernel<<<cvtBlocks, 256, 0, stream>>>(Wp, wpb, (int)NE_DD);
  cvt_bf16_kernel<<<cvtBlocks, 256, 0, stream>>>(Wo, wob, (int)NE_DD);
  cvt_bf16_kernel<<<cvtBlocks, 256, 0, stream>>>(posemb, peb, (int)(CT * CD));

  // 2) LayerNorm -> bf16
  ln_bf16_kernel<<<CB * CT, 256, 0, stream>>>(query, ln_g, ln_b, xb);

  // 3) projections (M=4096 for q/k/v, M=1024 for p)
  gemm_xw_kernel<EPI_QUQV><<<dim3(8, 32), 256, 0, stream>>>(
      xb, wqb, bq, qu, qv, nullptr, p_u, p_v);
  gemm_xw_kernel<EPI_HEADS><<<dim3(8, 32), 256, 0, stream>>>(
      xb, wkb, bk, kh, nullptr, nullptr, nullptr, nullptr);
  gemm_xw_kernel<EPI_HEADS><<<dim3(8, 32), 256, 0, stream>>>(
      xb, wvb, bv, vh, nullptr, nullptr, nullptr, nullptr);
  gemm_xw_kernel<EPI_PH><<<dim3(8, 8), 256, 0, stream>>>(
      peb, wpb, bp, ph, nullptr, nullptr, nullptr, nullptr);

  // 4) scores: ac -> attn region of d_out, bd -> workspace
  gemm_scores_kernel<0><<<dim3(8, 8, CB * CH), 256, 0, stream>>>(qu, kh, attn_f);
  gemm_scores_kernel<1><<<dim3(8, 8, CB * CH), 256, 0, stream>>>(qv, ph, bdb);

  // 5) rel-shift + scale + mask + softmax (in place on attn)
  softmax_shift_kernel<<<dim3(CT, CB * CH), 256, 0, stream>>>(attn_f, bdb, mask);

  // 6) attn @ V -> merged heads bf16
  gemm_av_kernel<<<dim3(1, 8, CB * CH), 256, 0, stream>>>(attn_f, vh, ao);

  // 7) output projection -> fp32 d_out
  gemm_xw_kernel<EPI_OUTPROJ><<<dim3(8, 32), 256, 0, stream>>>(
      ao, wob, bo, nullptr, nullptr, out_f, nullptr, nullptr);
}
